// Decoder_7705171329570
// MI455X (gfx1250) — matrix operand order repaired and checked by reading, not compiled, not run on hardware
//
#include <hip/hip_runtime.h>
#include <hip/hip_bf16.h>

// ---------------- problem constants ----------------
static constexpr int B_ = 512, S_ = 64, T_ = 32, V_ = 128;
static constexpr int E_ = 256, H_ = 512, DENC_ = 1024;
static constexpr int FOURH_ = 4 * H_;          // 2048
static constexpr int KX0_ = E_ + DENC_;        // 1280
static constexpr int KOUT_ = H_ + DENC_ + E_;  // 1792
static constexpr int EOS_ = 2;

typedef __bf16 bf16_t;
typedef __attribute__((ext_vector_type(16))) __bf16 v16bf;
typedef __attribute__((ext_vector_type(8)))  float  v8f;

// float -> bf16 round-to-nearest-even
__device__ __forceinline__ bf16_t f2b(float f) {
    unsigned u = __builtin_bit_cast(unsigned, f);
    unsigned r = (u + 0x7FFFu + ((u >> 16) & 1u)) >> 16;
    unsigned short s = (unsigned short)r;
    return __builtin_bit_cast(bf16_t, s);
}

// ---------------- CDNA5 async global->LDS (ASYNCcnt-tracked) ----------------
__device__ __forceinline__ void async_b128(unsigned lds_off, const void* gaddr) {
    asm volatile("global_load_async_to_lds_b128 %0, %1, off"
                 :: "v"(lds_off), "v"((unsigned long long)gaddr)
                 : "memory");
}
__device__ __forceinline__ void wait_asynccnt0() {
    asm volatile("s_wait_asynccnt 0x0" ::: "memory");
}

// ---------------- WMMA fragment helpers (16x16x32 bf16) ---------------------
// A (16x32 MxK) from bf16 global:
// lane<16:  row m0+lane,    K=[k0..k0+7] and [k0+16..k0+23]
// lane>=16: row m0+lane-16, K=[k0+8..k0+15] and [k0+24..k0+31]
__device__ __forceinline__ v16bf load_a_bf(const bf16_t* __restrict__ A, int lda,
                                           int m0, int k0, int lm, int halfsel) {
    const bf16_t* p = A + (size_t)(m0 + lm) * (size_t)lda + (size_t)(k0 + halfsel * 8);
    union { uint4 u[2]; v16bf v; } t;
    t.u[0] = *(const uint4*)(p);
    t.u[1] = *(const uint4*)(p + 16);
    return t.v;
}

// B (32x16 KxN = W^T) fragment from LDS tile sB[64 rows][32 k] bf16 row-major:
// lane<16: row n0t+lane, K=[0..7],[16..23]; lane>=16: row n0t+lane-16, K=[8..15],[24..31]
__device__ __forceinline__ v16bf load_b_lds(const bf16_t* sB, int trow,
                                            int lm, int halfsel) {
    const bf16_t* q = sB + (size_t)(trow + lm) * 32 + halfsel * 8;
    union { uint4 u[2]; v16bf v; } t;
    t.u[0] = *(const uint4*)(q);
    t.u[1] = *(const uint4*)(q + 16);
    return t.v;
}

// One K-segment of the GEMM: C += A(bf16) * W(bf16 NxK)^T with LDS-staged B
__device__ __forceinline__ void gemm_seg(
    const bf16_t* __restrict__ A, int K, const bf16_t* __restrict__ W,
    bf16_t* sB, unsigned sB_lds, int m0, int n0, int lm, int halfsel,
    int tid, v8f acc[4]) {
    const int brow = tid >> 2;   // 0..63 : row of 64-wide N tile
    const int bchk = tid & 3;    // 0..3  : 16B chunk within 64B row
    for (int k0 = 0; k0 < K; k0 += 32) {
        __syncthreads();  // previous tile fully consumed
        async_b128(sB_lds + (unsigned)(brow * 64 + bchk * 16),
                   (const void*)(W + (size_t)(n0 + brow) * (size_t)K + k0 + bchk * 8));
        v16bf a = load_a_bf(A, K, m0, k0, lm, halfsel);
        wait_asynccnt0();
        __syncthreads();  // tile visible to all waves
#pragma unroll
        for (int t = 0; t < 4; ++t) {
            v16bf b = load_b_lds(sB, 16 * t, lm, halfsel);
            acc[t] = __builtin_amdgcn_wmma_f32_16x16x32_bf16(
                false, a, false, b, (short)0, acc[t], false, false);
        }
    }
}

// ---------------- dual-K WMMA GEMM ------------------------------------------
// C[m,n] = bias[n] + sum A1[m,k]W1[n,k] (+ sum A2[m,k]W2[n,k] if A2)
// Block 256 thr = 8 waves; wave tile 16x64; block tile 128x64.
// grid.x = N/64, grid.y = M/128 (exact for all shapes here)
__global__ void __launch_bounds__(256) wmma_gemm_kernel(
    const bf16_t* __restrict__ A1, int K1, const bf16_t* __restrict__ W1,
    const bf16_t* __restrict__ A2, int K2, const bf16_t* __restrict__ W2,
    const float* __restrict__ bias,
    float* __restrict__ C, long long ldc) {
    __shared__ __align__(16) bf16_t sB[64 * 32];
    const int tid     = threadIdx.x;
    const int lane    = tid & 31;
    const int wave    = tid >> 5;
    const int lm      = lane & 15;
    const int halfsel = lane >> 4;
    const int m0 = blockIdx.y * 128 + wave * 16;
    const int n0 = blockIdx.x * 64;
    const unsigned sB_lds = (unsigned)(size_t)(void*)&sB[0];

    v8f acc[4];
#pragma unroll
    for (int t = 0; t < 4; ++t) {
        float bv = bias ? bias[n0 + 16 * t + lm] : 0.0f;
#pragma unroll
        for (int r = 0; r < 8; ++r) acc[t][r] = bv;
    }

    gemm_seg(A1, K1, W1, sB, sB_lds, m0, n0, lm, halfsel, tid, acc);
    if (A2) gemm_seg(A2, K2, W2, sB, sB_lds, m0, n0, lm, halfsel, tid, acc);

    // C/D: VGPR r -> row m0 + r + 8*halfsel, col n0 + 16t + lm
    const long long crow = (long long)(m0 + halfsel * 8);
#pragma unroll
    for (int t = 0; t < 4; ++t) {
        float* cp = C + crow * ldc + (long long)(n0 + 16 * t + lm);
#pragma unroll
        for (int r = 0; r < 8; ++r) cp[(long long)r * ldc] = acc[t][r];
    }
}

// ---------------- prep kernels ----------------------------------------------
__global__ void __launch_bounds__(256) f2bf_kernel(const float* __restrict__ s,
                                                   bf16_t* __restrict__ d, int n) {
    int i = blockIdx.x * 256 + threadIdx.x;
    if (i < n) d[i] = f2b(s[i]);
}

__global__ void __launch_bounds__(256) addvec_kernel(const float* __restrict__ a,
                                                     const float* __restrict__ b,
                                                     float* __restrict__ o, int n) {
    int i = blockIdx.x * 256 + threadIdx.x;
    if (i < n) o[i] = a[i] + b[i];
}

// ---------------- attention + concat (one block per batch row) --------------
__global__ void __launch_bounds__(256) attn_kernel(
    const float* __restrict__ enc_proj,     // (B*S, H) f32
    const float* __restrict__ encoder_out,  // (B, S, DENC) f32
    const float* __restrict__ dec_proj,     // (B, H) f32
    const float* __restrict__ vvec,         // (H)
    const int* __restrict__ mask,           // (B, S)
    const int* __restrict__ targets,        // (B, T)
    const float* __restrict__ emb,          // (V, E)
    int t,
    bf16_t* __restrict__ xbuf,              // (B, KX0)  bf16
    bf16_t* __restrict__ catbuf) {          // (B, KOUT) bf16
    const int b    = blockIdx.x;
    const int tid  = threadIdx.x;
    const int lane = tid & 31;
    const int wave = tid >> 5;

    __shared__ float s_score[S_];
    __shared__ float s_w[S_];

    const float* dp = dec_proj + (size_t)b * H_;
#pragma unroll
    for (int si = 0; si < 8; ++si) {
        int s = wave * 8 + si;
        const float* ep = enc_proj + ((size_t)b * S_ + s) * (size_t)H_;
        float part = 0.0f;
        for (int h = lane; h < H_; h += 32)
            part += vvec[h] * tanhf(dp[h] + ep[h]);
#pragma unroll
        for (int off = 16; off > 0; off >>= 1) part += __shfl_xor(part, off, 32);
        if (lane == 0) s_score[s] = part;
    }
    __syncthreads();

    if (wave == 0) {  // masked softmax over 64 scores (2 per lane)
        bool m0v = mask[b * S_ + lane] != 0;
        bool m1v = mask[b * S_ + lane + 32] != 0;
        float sc0 = m0v ? s_score[lane]      : -__builtin_inff();
        float sc1 = m1v ? s_score[lane + 32] : -__builtin_inff();
        float mx = fmaxf(sc0, sc1);
#pragma unroll
        for (int off = 16; off > 0; off >>= 1) mx = fmaxf(mx, __shfl_xor(mx, off, 32));
        float e0 = m0v ? __expf(sc0 - mx) : 0.0f;
        float e1 = m1v ? __expf(sc1 - mx) : 0.0f;
        float sum = e0 + e1;
#pragma unroll
        for (int off = 16; off > 0; off >>= 1) sum += __shfl_xor(sum, off, 32);
        float inv = 1.0f / sum;
        s_w[lane]      = e0 * inv;
        s_w[lane + 32] = e1 * inv;
    }
    __syncthreads();

    // context -> bf16 into x[:, E:] and cat[:, H:H+DENC]
    const float* eo = encoder_out + (size_t)b * S_ * DENC_;
    for (int d = tid; d < DENC_; d += 256) {
        float acc = 0.0f;
#pragma unroll 8
        for (int s = 0; s < S_; ++s) acc += s_w[s] * eo[(size_t)s * DENC_ + d];
        bf16_t cb = f2b(acc);
        xbuf[(size_t)b * KX0_ + E_ + d]    = cb;
        catbuf[(size_t)b * KOUT_ + H_ + d] = cb;
    }

    // embedding (teacher forcing): tok = t==0 ? EOS : targets[t-1]
    int tok = (t == 0) ? EOS_ : targets[b * T_ + (t - 1)];
    bf16_t ev = f2b(emb[(size_t)tok * E_ + tid]);  // E_ == blockDim.x
    xbuf[(size_t)b * KX0_ + tid]                 = ev;
    catbuf[(size_t)b * KOUT_ + H_ + DENC_ + tid] = ev;
}

// ---------------- LSTM pointwise --------------------------------------------
__global__ void __launch_bounds__(256) lstm_pointwise_kernel(
    const float* __restrict__ gates,   // (B, 4H) order i,f,g,o
    bf16_t* __restrict__ hbuf,         // (B, H) bf16 out
    float* __restrict__ cbuf,          // (B, H) f32 in/out
    bf16_t* __restrict__ cat_or_null) {// cat[:, 0:H] gets h_new (layer 1 only)
    int idx = blockIdx.x * 256 + threadIdx.x;  // over B*H
    int b = idx >> 9;
    int h = idx & (H_ - 1);
    const float* g = gates + (size_t)b * FOURH_;
    float gi = g[h], gf = g[H_ + h], gg = g[2 * H_ + h], go = g[3 * H_ + h];
    float si = 1.0f / (1.0f + __expf(-gi));
    float sf = 1.0f / (1.0f + __expf(-gf));
    float so = 1.0f / (1.0f + __expf(-go));
    float cn = sf * cbuf[idx] + si * tanhf(gg);
    float hn = so * tanhf(cn);
    cbuf[idx] = cn;
    bf16_t hb = f2b(hn);
    hbuf[idx] = hb;
    if (cat_or_null) cat_or_null[(size_t)b * KOUT_ + h] = hb;
}

// ---------------- host side --------------------------------------------------
extern "C" void kernel_launch(void* const* d_in, const int* in_sizes, int n_in,
                              void* d_out, int out_size, void* d_ws, size_t ws_size,
                              hipStream_t stream) {
    const float* encoder_out = (const float*)d_in[0];
    const float* h0          = (const float*)d_in[1];
    const float* c0          = (const float*)d_in[2];
    const int*   targets     = (const int*)d_in[3];
    const int*   mask        = (const int*)d_in[4];
    const float* emb         = (const float*)d_in[5];
    const float* W_enc       = (const float*)d_in[6];
    const float* W_dec       = (const float*)d_in[7];
    const float* vvec        = (const float*)d_in[8];
    const float* W_ih0       = (const float*)d_in[9];
    const float* W_hh0       = (const float*)d_in[10];
    const float* b_ih0       = (const float*)d_in[11];
    const float* b_hh0       = (const float*)d_in[12];
    const float* W_ih1       = (const float*)d_in[13];
    const float* W_hh1       = (const float*)d_in[14];
    const float* b_ih1       = (const float*)d_in[15];
    const float* b_hh1       = (const float*)d_in[16];
    const float* W_out       = (const float*)d_in[17];
    const float* b_out       = (const float*)d_in[18];
    float* out = (float*)d_out;

    char* p = (char*)d_ws;
    auto carve = [&](size_t bytes) {
        void* r = (void*)p;
        p += (bytes + 255) & ~(size_t)255;
        return r;
    };
    bf16_t* W_enc_bf = (bf16_t*)carve((size_t)H_ * DENC_ * 2);
    bf16_t* W_dec_bf = (bf16_t*)carve((size_t)H_ * H_ * 2);
    bf16_t* W_ih0_bf = (bf16_t*)carve((size_t)FOURH_ * KX0_ * 2);
    bf16_t* W_hh0_bf = (bf16_t*)carve((size_t)FOURH_ * H_ * 2);
    bf16_t* W_ih1_bf = (bf16_t*)carve((size_t)FOURH_ * H_ * 2);
    bf16_t* W_hh1_bf = (bf16_t*)carve((size_t)FOURH_ * H_ * 2);
    bf16_t* W_out_bf = (bf16_t*)carve((size_t)V_ * KOUT_ * 2);
    bf16_t* enc_bf   = (bf16_t*)carve((size_t)B_ * S_ * DENC_ * 2);
    float*  bias0    = (float*)carve((size_t)FOURH_ * 4);
    float*  bias1    = (float*)carve((size_t)FOURH_ * 4);
    float*  enc_proj = (float*)carve((size_t)B_ * S_ * H_ * 4);
    bf16_t* h_ws     = (bf16_t*)carve((size_t)2 * B_ * H_ * 2);
    float*  c_ws     = (float*)carve((size_t)2 * B_ * H_ * 4);
    float*  dec_proj = (float*)carve((size_t)B_ * H_ * 4);
    bf16_t* xbuf     = (bf16_t*)carve((size_t)B_ * KX0_ * 2);
    bf16_t* catbuf   = (bf16_t*)carve((size_t)B_ * KOUT_ * 2);
    float*  gates    = (float*)carve((size_t)B_ * FOURH_ * 4);

    bf16_t* h_l0 = h_ws;
    bf16_t* h_l1 = h_ws + (size_t)B_ * H_;
    float*  c_l0 = c_ws;
    float*  c_l1 = c_ws + (size_t)B_ * H_;

    // ---- one-time prep: conversions, bias sums, state init ----
    auto cvt = [&](const float* s, bf16_t* d, int n) {
        f2bf_kernel<<<(n + 255) / 256, 256, 0, stream>>>(s, d, n);
    };
    cvt(W_enc, W_enc_bf, H_ * DENC_);
    cvt(W_dec, W_dec_bf, H_ * H_);
    cvt(W_ih0, W_ih0_bf, FOURH_ * KX0_);
    cvt(W_hh0, W_hh0_bf, FOURH_ * H_);
    cvt(W_ih1, W_ih1_bf, FOURH_ * H_);
    cvt(W_hh1, W_hh1_bf, FOURH_ * H_);
    cvt(W_out, W_out_bf, V_ * KOUT_);
    cvt(encoder_out, enc_bf, B_ * S_ * DENC_);
    cvt(h0, h_ws, 2 * B_ * H_);
    addvec_kernel<<<FOURH_ / 256, 256, 0, stream>>>(b_ih0, b_hh0, bias0, FOURH_);
    addvec_kernel<<<FOURH_ / 256, 256, 0, stream>>>(b_ih1, b_hh1, bias1, FOURH_);
    hipMemcpyAsync(c_ws, c0, (size_t)2 * B_ * H_ * 4, hipMemcpyDeviceToDevice, stream);

    // ---- enc_proj = encoder_out (B*S,DENC) x W_enc^T -> (B*S,H) f32 ----
    wmma_gemm_kernel<<<dim3(H_ / 64, (B_ * S_) / 128), 256, 0, stream>>>(
        enc_bf, DENC_, W_enc_bf, nullptr, 0, nullptr, nullptr,
        enc_proj, (long long)H_);

    // ---- T sequential decode steps ----
    for (int t = 0; t < T_; ++t) {
        // dec_proj = h_top @ W_dec^T
        wmma_gemm_kernel<<<dim3(H_ / 64, B_ / 128), 256, 0, stream>>>(
            h_l1, H_, W_dec_bf, nullptr, 0, nullptr, nullptr,
            dec_proj, (long long)H_);

        // attention + x=[emb|ctx], cat[H:] = [ctx|emb]
        attn_kernel<<<B_, 256, 0, stream>>>(enc_proj, encoder_out, dec_proj,
                                            vvec, mask, targets, emb, t,
                                            xbuf, catbuf);

        // layer-0 gates = x@W_ih0^T + h0@W_hh0^T + (b_ih0+b_hh0)
        wmma_gemm_kernel<<<dim3(FOURH_ / 64, B_ / 128), 256, 0, stream>>>(
            xbuf, KX0_, W_ih0_bf, h_l0, H_, W_hh0_bf, bias0,
            gates, (long long)FOURH_);
        lstm_pointwise_kernel<<<(B_ * H_) / 256, 256, 0, stream>>>(
            gates, h_l0, c_l0, nullptr);

        // layer-1 gates = h0n@W_ih1^T + h1@W_hh1^T + (b_ih1+b_hh1)
        wmma_gemm_kernel<<<dim3(FOURH_ / 64, B_ / 128), 256, 0, stream>>>(
            h_l0, H_, W_ih1_bf, h_l1, H_, W_hh1_bf, bias1,
            gates, (long long)FOURH_);
        lstm_pointwise_kernel<<<(B_ * H_) / 256, 256, 0, stream>>>(
            gates, h_l1, c_l1, catbuf);

        // pred[:, t, :] = cat @ W_out^T + b_out  (strided into d_out)
        wmma_gemm_kernel<<<dim3(V_ / 64, B_ / 128), 256, 0, stream>>>(
            catbuf, KOUT_, W_out_bf, nullptr, 0, nullptr, b_out,
            out + (size_t)t * V_, (long long)(T_ * V_));
    }
    (void)in_sizes; (void)n_in; (void)out_size; (void)ws_size;
}
